// TransformerBlock_48352741818401
// MI455X (gfx1250) — matrix operand order repaired and checked by reading, not compiled, not run on hardware
//
#include <hip/hip_runtime.h>
#include <hip/hip_bf16.h>

typedef unsigned int   u32;
typedef unsigned short u16;

typedef __attribute__((ext_vector_type(16))) __bf16 v16bf;
typedef __attribute__((ext_vector_type(8)))  float  v8f;
typedef __attribute__((ext_vector_type(4)))  int    v4i;

#define DIM 2048
#define NH  16
#define HD  128
#define B_  2
#define S_  2048
#define HID 5632
#define MTOT (B_ * S_)          // 4096 token rows

#define AS1 __attribute__((address_space(1)))
#define AS3 __attribute__((address_space(3)))

#if __has_builtin(__builtin_amdgcn_global_load_async_to_lds_b128) && \
    __has_builtin(__builtin_amdgcn_s_wait_asynccnt)
#define USE_ASYNC_LDS 1
#else
#define USE_ASYNC_LDS 0
#endif

// ---------------------------------------------------------------------------
// helpers
// ---------------------------------------------------------------------------
__device__ __forceinline__ u16 f2bf(float f) {
    __hip_bfloat16 h = __float2bfloat16(f);
    u16 r; __builtin_memcpy(&r, &h, 2); return r;
}

union Frag { v16bf v; u32 u[8]; };

#if USE_ASYNC_LDS
__device__ __forceinline__ void async_b128(const u16* src, u16* dst) {
    __builtin_amdgcn_global_load_async_to_lds_b128(
        (AS1 v4i*)src, (AS3 v4i*)dst, 0, 0);
}
#endif

// ---------------------------------------------------------------------------
// fp32 -> bf16 elementwise convert (weights)
// ---------------------------------------------------------------------------
__global__ void f32_to_bf16_kernel(const float* __restrict__ in,
                                   u16* __restrict__ out, int n) {
    int i = blockIdx.x * blockDim.x + threadIdx.x;
    if (i < n) out[i] = f2bf(in[i]);
}

// ---------------------------------------------------------------------------
// RMSNorm: one block per row of 2048, bf16 output
// ---------------------------------------------------------------------------
__global__ __launch_bounds__(256) void rmsnorm_kernel(
    const float* __restrict__ x, const float* __restrict__ w,
    u16* __restrict__ out) {
    const int row = blockIdx.x;
    const float* xr = x + (size_t)row * DIM;
    float s = 0.f;
    for (int i = threadIdx.x; i < DIM; i += 256) { float v = xr[i]; s += v * v; }
#pragma unroll
    for (int off = 16; off >= 1; off >>= 1) s += __shfl_xor(s, off, 32);
    __shared__ float red[8];
    if ((threadIdx.x & 31) == 0) red[threadIdx.x >> 5] = s;
    __syncthreads();
    float tot = red[0] + red[1] + red[2] + red[3] + red[4] + red[5] + red[6] + red[7];
    float rs = rsqrtf(tot * (1.0f / DIM) + 1e-6f);
    u16* orow = out + (size_t)row * DIM;
    for (int i = threadIdx.x; i < DIM; i += 256) orow[i] = f2bf(xr[i] * rs * w[i]);
}

// ---------------------------------------------------------------------------
// Tiled bf16 WMMA GEMM: C(MxN,f32) = A(MxK,bf16) x B(KxN,bf16) [+ Res]
// block = 256 thr (8 waves), tile 128x128, BK=32
// wave (wm,wn) in 2x4 grid computes 64x32 via 4x2 v_wmma_f32_16x16x32_bf16
// A tile staged with GLOBAL_LOAD_ASYNC_TO_LDS when available (ASYNCcnt path)
// ---------------------------------------------------------------------------
#define GBM 128
#define GBN 128
#define GBK 32
#define LDSA 40   // A-tile LDS stride (bf16 elems): 80 B rows, 16 B aligned
#define LDSB 36   // transposed B-tile LDS stride

template <bool HasRes>
__global__ __launch_bounds__(256) void gemm_bf16_kernel(
    const u16* __restrict__ A, const u16* __restrict__ Bm,
    float* __restrict__ C, const float* __restrict__ Res,
    int M, int N, int K) {
    __shared__ u16 As[GBM * LDSA];   // row-major A tile rows
    __shared__ u16 Bs[GBN * LDSB];   // transposed B tile: [n][k]

    const int tid   = threadIdx.x;
    const int wave  = tid >> 5;
    const int lane  = tid & 31;
    const int lm    = lane & 15;
    const int lhalf = lane >> 4;
    const int wm    = wave >> 2;      // 0..1
    const int wn    = wave & 3;       // 0..3
    const int bm0   = blockIdx.y * GBM;
    const int bn0   = blockIdx.x * GBN;

    v8f acc[4][2];
#pragma unroll
    for (int i = 0; i < 4; ++i)
#pragma unroll
        for (int j = 0; j < 2; ++j) acc[i][j] = v8f{0.f,0.f,0.f,0.f,0.f,0.f,0.f,0.f};

    const int ar = tid >> 1;           // A tile row 0..127
    const int ac = (tid & 1) * 16;     // A tile col group
    const int br = tid >> 3;           // B tile k-row 0..31
    const int bc = (tid & 7) * 16;     // B tile col group
    const int kb = lhalf * 8;          // A-fragment K interleave base

    for (int k0 = 0; k0 < K; k0 += GBK) {
        // ---- stage A tile (row-major) ----
        {
            const u16* src = A + (size_t)(bm0 + ar) * K + k0 + ac;
            u16* dst = &As[ar * LDSA + ac];
#if USE_ASYNC_LDS
            async_b128(src, dst);
            async_b128(src + 8, dst + 8);
#else
            const uint4* s4 = (const uint4*)src;
            uint4 p0 = s4[0], p1 = s4[1];
            u32* d4 = (u32*)dst;
            d4[0]=p0.x; d4[1]=p0.y; d4[2]=p0.z; d4[3]=p0.w;
            d4[4]=p1.x; d4[5]=p1.y; d4[6]=p1.z; d4[7]=p1.w;
#endif
        }
        // ---- stage B tile transposed: Bs[n][k] ----
        {
            const uint4* src = (const uint4*)(Bm + (size_t)(k0 + br) * N + bn0 + bc);
            uint4 p0 = src[0], p1 = src[1];
            u16 tmp[16];
            *(uint4*)&tmp[0] = p0; *(uint4*)&tmp[8] = p1;
#pragma unroll
            for (int j = 0; j < 16; ++j) Bs[(bc + j) * LDSB + br] = tmp[j];
        }
#if USE_ASYNC_LDS
        __builtin_amdgcn_s_wait_asynccnt(0);
#endif
        __syncthreads();

        Frag a[4], b[2];
#pragma unroll
        for (int mf = 0; mf < 4; ++mf) {
            const u32* p = (const u32*)&As[(wm * 64 + mf * 16 + lm) * LDSA];
#pragma unroll
            for (int j = 0; j < 4; ++j) { a[mf].u[j] = p[(kb >> 1) + j]; a[mf].u[4 + j] = p[8 + (kb >> 1) + j]; }
        }
#pragma unroll
        for (int nf = 0; nf < 2; ++nf) {
            const u32* p = (const u32*)&Bs[(wn * 32 + nf * 16 + lm) * LDSB];
#pragma unroll
            for (int j = 0; j < 8; ++j) b[nf].u[j] = p[(j >> 2) * 8 + lhalf * 4 + (j & 3)];
        }
#pragma unroll
        for (int mf = 0; mf < 4; ++mf)
#pragma unroll
            for (int nf = 0; nf < 2; ++nf)
                acc[mf][nf] = __builtin_amdgcn_wmma_f32_16x16x32_bf16(
                    false, a[mf].v, false, b[nf].v, (short)0, acc[mf][nf], false, false);
        __syncthreads();
    }

#pragma unroll
    for (int mf = 0; mf < 4; ++mf)
#pragma unroll
        for (int nf = 0; nf < 2; ++nf)
#pragma unroll
            for (int r = 0; r < 8; ++r) {
                int row = bm0 + wm * 64 + mf * 16 + r + 8 * lhalf;
                int col = bn0 + wn * 32 + nf * 16 + lm;
                size_t idx = (size_t)row * N + col;
                float v = acc[mf][nf][r];
                if (HasRes) v += Res[idx];
                C[idx] = v;
            }
}

// ---------------------------------------------------------------------------
// RoPE on q,k (f32 in) -> bf16 out; v f32 -> bf16 passthrough
// ---------------------------------------------------------------------------
__global__ void rope_qkv_kernel(const float* __restrict__ q,
                                const float* __restrict__ k,
                                const float* __restrict__ v,
                                const float* __restrict__ fc,
                                const float* __restrict__ fs,
                                u16* __restrict__ qb, u16* __restrict__ kb,
                                u16* __restrict__ vb) {
    int p = blockIdx.x * blockDim.x + threadIdx.x;       // 0 .. MTOT*1024-1
    if (p >= MTOT * (DIM / 2)) return;
    int t  = p >> 10;                 // token row
    int pr = p & 1023;
    int h  = pr >> 6;                 // head
    int d2 = pr & 63;                 // pair index in head
    int s  = t & (S_ - 1);
    float c = fc[s * (HD / 2) + d2];
    float sn = fs[s * (HD / 2) + d2];
    size_t idx = (size_t)t * DIM + h * HD + 2 * d2;
    float qre = q[idx], qim = q[idx + 1];
    float kre = k[idx], kim = k[idx + 1];
    qb[idx]     = f2bf(qre * c - qim * sn);
    qb[idx + 1] = f2bf(qre * sn + qim * c);
    kb[idx]     = f2bf(kre * c - kim * sn);
    kb[idx + 1] = f2bf(kre * sn + kim * c);
    vb[idx]     = f2bf(v[idx]);
    vb[idx + 1] = f2bf(v[idx + 1]);
}

// ---------------------------------------------------------------------------
// Flash attention, causal. block = 128 thr (4 waves), each wave = 16 q rows,
// q-tile = 64 rows, k-tile = 32 rows. QK^T and PV via bf16 WMMA, fp32 softmax.
// K tile staged via async-to-LDS when available.
// ---------------------------------------------------------------------------
#define KST 136   // K-tile LDS stride (272 B rows, 16 B aligned)

__global__ __launch_bounds__(128) void flash_attn_kernel(
    const u16* __restrict__ Qb, const u16* __restrict__ Kb,
    const u16* __restrict__ Vb, u16* __restrict__ Ob) {
    __shared__ u16 Ks[32 * KST];        // K tile row-major [krow][d]
    __shared__ u16 Vst[HD * LDSB];      // V tile col-major [d][krow]
    __shared__ u16 Ps[4][16 * LDSB];    // per-wave P scratch [m][kcol]

    const int tid   = threadIdx.x;
    const int wave  = tid >> 5;
    const int lane  = tid & 31;
    const int lm    = lane & 15;
    const int lhalf = lane >> 4;
    const int kbI   = lhalf * 8;        // A-fragment interleave base

    const int qt = blockIdx.x;  const int q0 = qt * 64;
    const int bh = blockIdx.y;  const int b = bh >> 4; const int h = bh & 15;

    // Q fragments (16 rows x 128 d) held in registers for the whole loop
    Frag aq[4];
    {
        const int qrow = q0 + wave * 16 + lm;
        const u32* qp = (const u32*)(Qb + (size_t)(b * S_ + qrow) * DIM + h * HD);
#pragma unroll
        for (int kk = 0; kk < 4; ++kk) {
#pragma unroll
            for (int j = 0; j < 4; ++j) {
                aq[kk].u[j]     = qp[kk * 16 + (kbI >> 1) + j];
                aq[kk].u[4 + j] = qp[kk * 16 + 8 + (kbI >> 1) + j];
            }
        }
    }

    float mrow[8], lrow[8];
    v8f o[8];
#pragma unroll
    for (int r = 0; r < 8; ++r) { mrow[r] = -1e30f; lrow[r] = 0.f; }
#pragma unroll
    for (int c = 0; c < 8; ++c) o[c] = v8f{0.f,0.f,0.f,0.f,0.f,0.f,0.f,0.f};

    const int ntiles = 2 * qt + 2;            // causal bound, block-uniform
    for (int kt = 0; kt < ntiles; ++kt) {
        __syncthreads();
        // ---- stage K tile (row-major, async) and V tile (col-major) ----
        {
            const int r = tid >> 2; const int cg = (tid & 3) * 32;
            const u16* ksrc = Kb + (size_t)(b * S_ + kt * 32 + r) * DIM + h * HD + cg;
            u16* kdst = &Ks[r * KST + cg];
#if USE_ASYNC_LDS
#pragma unroll
            for (int j = 0; j < 4; ++j)
                async_b128(ksrc + 8 * j, kdst + 8 * j);
#else
            const uint4* ks4 = (const uint4*)ksrc;
            uint4 kp0 = ks4[0], kp1 = ks4[1], kp2 = ks4[2], kp3 = ks4[3];
            u32* kd = (u32*)kdst;
            kd[0]=kp0.x; kd[1]=kp0.y; kd[2]=kp0.z; kd[3]=kp0.w;
            kd[4]=kp1.x; kd[5]=kp1.y; kd[6]=kp1.z; kd[7]=kp1.w;
            kd[8]=kp2.x; kd[9]=kp2.y; kd[10]=kp2.z; kd[11]=kp2.w;
            kd[12]=kp3.x; kd[13]=kp3.y; kd[14]=kp3.z; kd[15]=kp3.w;
#endif
            const uint4* vs = (const uint4*)(Vb + (size_t)(b * S_ + kt * 32 + r) * DIM + h * HD + cg);
            u16 tmp[32];
            *(uint4*)&tmp[0]  = vs[0]; *(uint4*)&tmp[8]  = vs[1];
            *(uint4*)&tmp[16] = vs[2]; *(uint4*)&tmp[24] = vs[3];
#pragma unroll
            for (int j = 0; j < 32; ++j) Vst[(cg + j) * LDSB + r] = tmp[j];
        }
#if USE_ASYNC_LDS
        __builtin_amdgcn_s_wait_asynccnt(0);
#endif
        __syncthreads();

        // ---- scores: S = Q * K^T (16x32), two 16x16 WMMA accumulators ----
        v8f sc[2];
        sc[0] = v8f{0.f,0.f,0.f,0.f,0.f,0.f,0.f,0.f};
        sc[1] = v8f{0.f,0.f,0.f,0.f,0.f,0.f,0.f,0.f};
#pragma unroll
        for (int nf = 0; nf < 2; ++nf) {
            const u32* kp = (const u32*)&Ks[(nf * 16 + lm) * KST];
#pragma unroll
            for (int kk = 0; kk < 4; ++kk) {
                Frag bk;
#pragma unroll
                for (int j = 0; j < 8; ++j) bk.u[j] = kp[kk * 16 + (j >> 2) * 8 + lhalf * 4 + (j & 3)];
                sc[nf] = __builtin_amdgcn_wmma_f32_16x16x32_bf16(
                    false, aq[kk].v, false, bk.v, (short)0, sc[nf], false, false);
            }
        }

        // ---- online softmax (per wave rows; row striped over 16 lanes) ----
        const float scale = 0.08838834764831845f;   // 1/sqrt(128)
        const int qrb = q0 + wave * 16;
#pragma unroll
        for (int r = 0; r < 8; ++r) {
            const int qrow = qrb + r + 8 * lhalf;
            float s0 = sc[0][r] * scale; if (kt * 32 + lm      > qrow) s0 = -1e30f;
            float s1 = sc[1][r] * scale; if (kt * 32 + 16 + lm > qrow) s1 = -1e30f;
            float t = fmaxf(s0, s1);
#pragma unroll
            for (int off = 8; off >= 1; off >>= 1) t = fmaxf(t, __shfl_xor(t, off, 32));
            float mnew = fmaxf(mrow[r], t);
            float corr = __expf(mrow[r] - mnew);
            mrow[r] = mnew;
            float p0 = __expf(s0 - mnew);
            float p1 = __expf(s1 - mnew);
            float rs = p0 + p1;
#pragma unroll
            for (int off = 8; off >= 1; off >>= 1) rs += __shfl_xor(rs, off, 32);
            lrow[r] = lrow[r] * corr + rs;
#pragma unroll
            for (int c = 0; c < 8; ++c) o[c][r] *= corr;
            u16* pp = &Ps[wave][(r + 8 * lhalf) * LDSB];
            pp[lm]      = f2bf(p0);
            pp[16 + lm] = f2bf(p1);
        }
        __syncthreads();

        // ---- P fragment (A-layout) + O += P * V ----
        Frag pf;
        {
            const u32* pp = (const u32*)&Ps[wave][lm * LDSB];
#pragma unroll
            for (int j = 0; j < 4; ++j) { pf.u[j] = pp[(kbI >> 1) + j]; pf.u[4 + j] = pp[8 + (kbI >> 1) + j]; }
        }
#pragma unroll
        for (int c = 0; c < 8; ++c) {
            Frag bv;
            const u32* vp = (const u32*)&Vst[(c * 16 + lm) * LDSB];
#pragma unroll
            for (int j = 0; j < 8; ++j) bv.u[j] = vp[(j >> 2) * 8 + lhalf * 4 + (j & 3)];
            o[c] = __builtin_amdgcn_wmma_f32_16x16x32_bf16(
                false, pf.v, false, bv.v, (short)0, o[c], false, false);
        }
    }

    // ---- finalize: O /= l, write bf16 ----
#pragma unroll
    for (int r = 0; r < 8; ++r) {
        float inv = 1.0f / lrow[r];
        int row = b * S_ + q0 + wave * 16 + r + 8 * lhalf;
        u16* dst = Ob + (size_t)row * DIM + h * HD;
#pragma unroll
        for (int c = 0; c < 8; ++c) dst[c * 16 + lm] = f2bf(o[c][r] * inv);
    }
}

// ---------------------------------------------------------------------------
// g = silu(t1) * t3  -> bf16
// ---------------------------------------------------------------------------
__global__ void silu_mul_kernel(const float* __restrict__ t1,
                                const float* __restrict__ t3,
                                u16* __restrict__ g, int n) {
    int i = blockIdx.x * blockDim.x + threadIdx.x;
    if (i < n) {
        float a = t1[i];
        float s = a / (1.0f + __expf(-a));
        g[i] = f2bf(s * t3[i]);
    }
}

// ---------------------------------------------------------------------------
// host launcher
// ---------------------------------------------------------------------------
extern "C" void kernel_launch(void* const* d_in, const int* in_sizes, int n_in,
                              void* d_out, int out_size, void* d_ws, size_t ws_size,
                              hipStream_t stream) {
    (void)in_sizes; (void)n_in; (void)out_size; (void)ws_size;
    const float* x    = (const float*)d_in[0];
    const float* wq   = (const float*)d_in[1];
    const float* wk   = (const float*)d_in[2];
    const float* wv   = (const float*)d_in[3];
    const float* wo   = (const float*)d_in[4];
    const float* w1   = (const float*)d_in[5];
    const float* w2   = (const float*)d_in[6];
    const float* w3   = (const float*)d_in[7];
    const float* anw  = (const float*)d_in[8];
    const float* fnw  = (const float*)d_in[9];
    const float* fc   = (const float*)d_in[10];
    const float* fs   = (const float*)d_in[11];
    float* out        = (float*)d_out;

    char* ws = (char*)d_ws;
    constexpr size_t SZ_W4     = (size_t)DIM * DIM * 2;
    constexpr size_t SZ_WH     = (size_t)DIM * HID * 2;
    constexpr size_t SZ_ACT_BF = (size_t)MTOT * DIM * 2;
    constexpr size_t SZ_ACT_F  = (size_t)MTOT * DIM * 4;
    constexpr size_t SZ_HID_F  = (size_t)MTOT * HID * 4;

    size_t off = 0;
    u16* wqb = (u16*)(ws + off); off += SZ_W4;
    u16* wkb = (u16*)(ws + off); off += SZ_W4;
    u16* wvb = (u16*)(ws + off); off += SZ_W4;
    u16* wob = (u16*)(ws + off); off += SZ_W4;
    u16* w1b = (u16*)(ws + off); off += SZ_WH;
    u16* w2b = (u16*)(ws + off); off += SZ_WH;
    u16* w3b = (u16*)(ws + off); off += SZ_WH;
    float* h = (float*)(ws + off); off += SZ_ACT_F;
    u16* hin = (u16*)(ws + off); off += SZ_ACT_BF;   // reused as fin
    // region C: q/k/v f32 (3x) reused later for t1 (SZ_HID_F <= 3x SZ_ACT_F)
    float* qf = (float*)(ws + off);
    float* kf = (float*)(ws + off + SZ_ACT_F);
    float* vf = (float*)(ws + off + 2 * SZ_ACT_F);
    float* t1 = qf;
    off += 3 * SZ_ACT_F;
    // region D: q/k/v bf16 (3x) reused later for t3 (reserve SZ_HID_F)
    u16* qb = (u16*)(ws + off);
    u16* kb = (u16*)(ws + off + SZ_ACT_BF);
    u16* vb = (u16*)(ws + off + 2 * SZ_ACT_BF);
    float* t3 = (float*)(ws + off);
    off += SZ_HID_F;
    // region F: attn bf16 reused later for g bf16
    u16* attnb = (u16*)(ws + off);
    u16* gb    = (u16*)(ws + off);

    // 1) weights -> bf16
    {
        int n4 = DIM * DIM, nh = DIM * HID;
        f32_to_bf16_kernel<<<(n4 + 255) / 256, 256, 0, stream>>>(wq, wqb, n4);
        f32_to_bf16_kernel<<<(n4 + 255) / 256, 256, 0, stream>>>(wk, wkb, n4);
        f32_to_bf16_kernel<<<(n4 + 255) / 256, 256, 0, stream>>>(wv, wvb, n4);
        f32_to_bf16_kernel<<<(n4 + 255) / 256, 256, 0, stream>>>(wo, wob, n4);
        f32_to_bf16_kernel<<<(nh + 255) / 256, 256, 0, stream>>>(w1, w1b, nh);
        f32_to_bf16_kernel<<<(nh + 255) / 256, 256, 0, stream>>>(w2, w2b, nh);
        f32_to_bf16_kernel<<<(nh + 255) / 256, 256, 0, stream>>>(w3, w3b, nh);
    }
    // 2) attn rmsnorm
    rmsnorm_kernel<<<MTOT, 256, 0, stream>>>(x, anw, hin);
    // 3) q/k/v projections (f32 out)
    {
        dim3 grid(DIM / GBN, MTOT / GBM);
        gemm_bf16_kernel<false><<<grid, 256, 0, stream>>>(hin, wqb, qf, nullptr, MTOT, DIM, DIM);
        gemm_bf16_kernel<false><<<grid, 256, 0, stream>>>(hin, wkb, kf, nullptr, MTOT, DIM, DIM);
        gemm_bf16_kernel<false><<<grid, 256, 0, stream>>>(hin, wvb, vf, nullptr, MTOT, DIM, DIM);
    }
    // 4) rope + bf16 pack
    {
        int n = MTOT * (DIM / 2);
        rope_qkv_kernel<<<(n + 255) / 256, 256, 0, stream>>>(qf, kf, vf, fc, fs, qb, kb, vb);
    }
    // 5) flash attention
    flash_attn_kernel<<<dim3(S_ / 64, B_ * NH), 128, 0, stream>>>(qb, kb, vb, attnb);
    // 6) h = x + attn @ wo
    gemm_bf16_kernel<true><<<dim3(DIM / GBN, MTOT / GBM), 256, 0, stream>>>(
        attnb, wob, h, x, MTOT, DIM, DIM);
    // 7) ffn rmsnorm (reuse hin buffer)
    rmsnorm_kernel<<<MTOT, 256, 0, stream>>>(h, fnw, hin);
    // 8) t1 = fin @ w1 ; t3 = fin @ w3
    {
        dim3 grid(HID / GBN, MTOT / GBM);
        gemm_bf16_kernel<false><<<grid, 256, 0, stream>>>(hin, w1b, t1, nullptr, MTOT, HID, DIM);
        gemm_bf16_kernel<false><<<grid, 256, 0, stream>>>(hin, w3b, t3, nullptr, MTOT, HID, DIM);
    }
    // 9) g = silu(t1) * t3
    {
        int n = MTOT * HID;
        silu_mul_kernel<<<(n + 255) / 256, 256, 0, stream>>>(t1, t3, gb, n);
    }
    // 10) out = h + g @ w2
    gemm_bf16_kernel<true><<<dim3(DIM / GBN, MTOT / GBM), 256, 0, stream>>>(
        gb, w2b, out, h, MTOT, DIM, HID);
}
